// Model_57664230916704
// MI455X (gfx1250) — matrix operand order repaired and checked by reading, not compiled, not run on hardware
//
#include <hip/hip_runtime.h>
#include <hip/hip_bf16.h>

typedef __attribute__((ext_vector_type(16))) _Float16 v16h;
typedef __attribute__((ext_vector_type(8)))  _Float16 v8h;
typedef __attribute__((ext_vector_type(8)))  float    v8f;

#define B_   4096
#define T_   100
#define D_   64
#define H_   128
#define BM   16          // batch rows per workgroup (WMMA M)
#define AS   264         // Ash row stride in halves (256 + 8 pad -> conflict-free b128 reads)
#define DS_  72          // Dsh row stride (64 + 8)
#define XS_  72          // XHsh row stride

// ---------------- fragment helpers (CDNA5 wave32 WMMA layouts) ----------------

// A (16x32 f16): lane l holds row (l&15); element e -> K = 32*kc + (e>=8?16:0) + (l>=16?8:0) + (e&7)
// i.e. two contiguous 8-half (16B) chunks per lane -> two ds_read_b128.
__device__ __forceinline__ v16h load_a(const _Float16* base, int stride, int l, int kb) {
  const _Float16* p = base + (l & 15) * stride + kb + ((l & 16) ? 8 : 0);
  v8h lo = *(const v8h*)p;
  v8h hi = *(const v8h*)(p + 16);
  v16h a;
#pragma unroll
  for (int i = 0; i < 8; ++i) { a[i] = lo[i]; a[i + 8] = hi[i]; }
  return a;
}

// B (32x16 f16) pre-packed fragment-linear in workspace: frag stride 512 halves, lane stride 16 halves.
__device__ __forceinline__ v16h load_b(const _Float16* frags, int fidx, int l) {
  return *(const v16h*)(frags + (size_t)fidx * 512 + (size_t)l * 16);
}

__device__ __forceinline__ v8f wmma16(v16h a, v16h b, v8f c) {
  return __builtin_amdgcn_wmma_f32_16x16x32_f16(false, a, false, b, (short)0, c, false, false);
}

__device__ __forceinline__ float sigmoidf_(float x) { return 1.f / (1.f + __expf(-x)); }

// ---------------- pre-pass 1: per-step mask denominators + zero loss ----------------
__global__ void prep_den(const float* __restrict__ masks, float* __restrict__ den,
                         float* __restrict__ outLoss) {
  const int t = blockIdx.x;
  float s = 0.f;
  for (int idx = threadIdx.x; idx < B_ * D_; idx += 256) {
    int b = idx >> 6, d = idx & 63;
    s += masks[((size_t)b * T_ + t) * D_ + d];
  }
#pragma unroll
  for (int off = 16; off; off >>= 1) s += __shfl_xor(s, off, 32);
  __shared__ float red[8];
  if ((threadIdx.x & 31) == 0) red[threadIdx.x >> 5] = s;
  __syncthreads();
  if (threadIdx.x == 0) {
    float tot = 0.f;
#pragma unroll
    for (int i = 0; i < 8; ++i) tot += red[i];
    den[t] = tot;
    if (t == 0) *outLoss = 0.f;
  }
}

// ---------------- pre-pass 2: f32 weights -> f16 B-fragments in workspace ----------------
// Wc = [W_ih | W_hh]^T  : K=256, N=512 -> 8x32 = 256 frags
// Wg = W_gamma^T        : K=64,  N=128 -> 2x8  = 16 frags
// Wr = W_reg^T          : K=128, N=64  -> 4x4  = 16 frags
__global__ void prep_frags(const float* __restrict__ Wgamma, const float* __restrict__ Wreg,
                           const float* __restrict__ Wih,    const float* __restrict__ Whh,
                           _Float16* __restrict__ wsWc, _Float16* __restrict__ wsWg,
                           _Float16* __restrict__ wsWr) {
  const int f = blockIdx.x;     // 0..287
  const int l = threadIdx.x;    // 0..31
  const int l15 = l & 15;
  const int kh  = (l & 16) ? 16 : 0;
  _Float16* dst; int kc, nt, mode;
  if (f < 256)      { mode = 0; kc = f >> 5;          nt = f & 31;         dst = wsWc + (size_t)f * 512; }
  else if (f < 272) { mode = 1; int fi = f - 256; kc = fi >> 3; nt = fi & 7; dst = wsWg + (size_t)fi * 512; }
  else              { mode = 2; int fi = f - 272; kc = fi >> 2; nt = fi & 3; dst = wsWr + (size_t)fi * 512; }
  const int n = nt * 16 + l15;
#pragma unroll
  for (int e = 0; e < 16; ++e) {
    const int k = kc * 32 + (kh >> 1) + (e & 7) + ((e & 8) << 1);
    float v;
    if (mode == 0)      v = (k < 128) ? Wih[(size_t)n * 128 + k] : Whh[(size_t)n * 128 + (k - 128)];
    else if (mode == 1) v = Wgamma[(size_t)n * 64 + k];
    else                v = Wreg[(size_t)n * 128 + k];
    dst[(size_t)l * 16 + e] = (_Float16)v;
  }
}

// ---------------- main recurrence: 256 WGs x 256 threads, h/c resident in registers ----------------
__global__ __launch_bounds__(256)
void brits_main(const float* __restrict__ values, const float* __restrict__ masks,
                const float* __restrict__ deltas,
                const float* __restrict__ b_gamma, const float* __restrict__ b_reg,
                const float* __restrict__ b_ih,    const float* __restrict__ b_hh,
                const float* __restrict__ den,
                const _Float16* __restrict__ Wc, const _Float16* __restrict__ Wg,
                const _Float16* __restrict__ Wr,
                float* __restrict__ outLoss, float* __restrict__ outImp) {
  // A-staging: cols 0..63 = x_c, 64..127 = m, 128..255 = decayed h (all f16, A-layout friendly)
  __shared__ __align__(32) _Float16 Ash[BM][AS];
  __shared__ __align__(32) _Float16 Dsh[BM][DS_];
  __shared__ __align__(16) float    XHsh[BM][XS_];
  __shared__ float lred[8];

  const int tid = threadIdx.x;
  const int wv  = tid >> 5;          // wave 0..7 -> owns h/c cols [16w,16w+16)
  const int l   = tid & 31;
  const int l15 = l & 15;
  const int lh8 = (l & 16) ? 8 : 0;  // D-fragment row offset
  const int r0  = blockIdx.x * BM;
  const int pr  = tid >> 4;          // pointwise mapping: row 0..15
  const int pc  = (tid & 15) * 4;    // 4 contiguous cols

  // per-lane biases (constant across t)
  const int nc = 16 * wv + l15;
  const float bias_g = b_gamma[nc];
  const float bias_r = (wv < 4) ? b_reg[nc] : 0.f;
  const float bias_i = b_ih[nc]       + b_hh[nc];
  const float bias_f = b_ih[128 + nc] + b_hh[128 + nc];
  const float bias_c = b_ih[256 + nc] + b_hh[256 + nc];
  const float bias_o = b_ih[384 + nc] + b_hh[384 + nc];

  v8f h, c;
#pragma unroll
  for (int e = 0; e < 8; ++e) { h[e] = 0.f; c[e] = 0.f; }
  float loss_local = 0.f;

  for (int t = 0; t < T_; ++t) {
    // -------- S0: global loads for this step (registers only) --------
    const size_t gbase = ((size_t)(r0 + pr) * T_ + t) * D_ + pc;
    const float4 xv = *(const float4*)(values + gbase);
    const float4 mv = *(const float4*)(masks  + gbase);
    const float4 dv = *(const float4*)(deltas + gbase);
    if (t + 1 < T_) {                       // gfx1250 global_prefetch_b8 for next step
      __builtin_prefetch(values + gbase + D_, 0, 0);
      __builtin_prefetch(masks  + gbase + D_, 0, 0);
      __builtin_prefetch(deltas + gbase + D_, 0, 0);
    }
    const float invden = 1.f / (den[t] + 1e-5f);

    __syncthreads();                        // B0: prior-step gate reads of Ash done
    Dsh[pr][pc + 0] = (_Float16)dv.x;  Dsh[pr][pc + 1] = (_Float16)dv.y;
    Dsh[pr][pc + 2] = (_Float16)dv.z;  Dsh[pr][pc + 3] = (_Float16)dv.w;
    Ash[pr][64 + pc + 0] = (_Float16)mv.x;  Ash[pr][64 + pc + 1] = (_Float16)mv.y;
    Ash[pr][64 + pc + 2] = (_Float16)mv.z;  Ash[pr][64 + pc + 3] = (_Float16)mv.w;
    __syncthreads();                        // B1: Dsh / m-part visible

    // -------- S1: gamma = exp(-relu(d@Wg^T+bg)); h *= gamma; stage h_f16 --------
    {
      v8f acc;
#pragma unroll
      for (int e = 0; e < 8; ++e) acc[e] = bias_g;
      acc = wmma16(load_a(&Dsh[0][0], DS_, l, 0),  load_b(Wg, 0 * 8 + wv, l), acc);
      acc = wmma16(load_a(&Dsh[0][0], DS_, l, 32), load_b(Wg, 1 * 8 + wv, l), acc);
#pragma unroll
      for (int e = 0; e < 8; ++e) h[e] *= __expf(-fmaxf(acc[e], 0.f));
#pragma unroll
      for (int e = 0; e < 8; ++e) Ash[e + lh8][128 + nc] = (_Float16)h[e];
    }
    __syncthreads();                        // B2: decayed h visible

    // -------- S2: x_h = h@Wr^T + br (waves 0..3 cover N=64) --------
    if (wv < 4) {
      v8f acc;
#pragma unroll
      for (int e = 0; e < 8; ++e) acc[e] = bias_r;
#pragma unroll
      for (int kc = 0; kc < 4; ++kc)
        acc = wmma16(load_a(&Ash[0][0], AS, l, 128 + 32 * kc), load_b(Wr, kc * 4 + wv, l), acc);
#pragma unroll
      for (int e = 0; e < 8; ++e) XHsh[e + lh8][16 * wv + l15] = acc[e];
    }
    __syncthreads();                        // B3: x_h visible

    // -------- S3: x_c = m*x + (1-m)*x_h; loss numerator; write imputations --------
    {
      const float xs[4] = { xv.x, xv.y, xv.z, xv.w };
      const float ms[4] = { mv.x, mv.y, mv.z, mv.w };
      float num = 0.f;
      float* op = outImp + gbase;
#pragma unroll
      for (int j = 0; j < 4; ++j) {
        const float xh = XHsh[pr][pc + j];
        const float m  = ms[j], x = xs[j];
        const float v  = m * x + (1.f - m) * xh;
        Ash[pr][pc + j] = (_Float16)v;
        op[j] = v;                          // outImp = d_out+1 -> keep scalar stores
        num += fabsf(x - xh) * m;
      }
      loss_local += num * invden;
    }
    __syncthreads();                        // B4: x_c part of Ash visible

    // -------- S4: gates = [x_c|m|h] @ [W_ih|W_hh]^T + biases; LSTM update --------
    {
      v8f ai, af, ag, ao;
#pragma unroll
      for (int e = 0; e < 8; ++e) { ai[e] = bias_i; af[e] = bias_f; ag[e] = bias_c; ao[e] = bias_o; }
#pragma unroll
      for (int kc = 0; kc < 8; ++kc) {
        const v16h a = load_a(&Ash[0][0], AS, l, 32 * kc);
        ai = wmma16(a, load_b(Wc, kc * 32 + wv,      l), ai);
        af = wmma16(a, load_b(Wc, kc * 32 + wv + 8,  l), af);
        ag = wmma16(a, load_b(Wc, kc * 32 + wv + 16, l), ag);
        ao = wmma16(a, load_b(Wc, kc * 32 + wv + 24, l), ao);
      }
#pragma unroll
      for (int e = 0; e < 8; ++e) {
        const float is = sigmoidf_(ai[e]);
        const float fs = sigmoidf_(af[e]);
        const float gt = tanhf(ag[e]);
        const float os = sigmoidf_(ao[e]);
        c[e] = fs * c[e] + is * gt;
        h[e] = os * tanhf(c[e]);
      }
    }
  }

  // -------- loss reduction: wave shuffle -> LDS -> one atomicAdd per WG --------
  float v = loss_local;
#pragma unroll
  for (int off = 16; off; off >>= 1) v += __shfl_xor(v, off, 32);
  if (l == 0) lred[wv] = v;
  __syncthreads();
  if (tid == 0) {
    float s = 0.f;
#pragma unroll
    for (int i = 0; i < 8; ++i) s += lred[i];
    atomicAdd(outLoss, s);
  }
}

// ---------------- host entry ----------------
extern "C" void kernel_launch(void* const* d_in, const int* in_sizes, int n_in,
                              void* d_out, int out_size, void* d_ws, size_t ws_size,
                              hipStream_t stream) {
  (void)in_sizes; (void)n_in; (void)out_size; (void)ws_size;
  const float* values = (const float*)d_in[0];
  const float* masks  = (const float*)d_in[1];
  const float* deltas = (const float*)d_in[2];
  const float* Wgamma = (const float*)d_in[3];
  const float* bgamma = (const float*)d_in[4];
  const float* Wreg   = (const float*)d_in[5];
  const float* breg   = (const float*)d_in[6];
  const float* Wih    = (const float*)d_in[7];
  const float* bih    = (const float*)d_in[8];
  const float* Whh    = (const float*)d_in[9];
  const float* bhh    = (const float*)d_in[10];

  float* out = (float*)d_out;               // out[0] = loss, out[1..] = imputations [B,T,D]
  char*  ws  = (char*)d_ws;
  float*    den  = (float*)ws;                                   // 100 f32
  _Float16* wsWc = (_Float16*)(ws + 1024);                       // 256 KB
  _Float16* wsWg = (_Float16*)(ws + 1024 + 256 * 1024);          // 16 KB
  _Float16* wsWr = (_Float16*)(ws + 1024 + 272 * 1024);          // 16 KB

  prep_den<<<T_, 256, 0, stream>>>(masks, den, out);
  prep_frags<<<288, 32, 0, stream>>>(Wgamma, Wreg, Wih, Whh, wsWc, wsWg, wsWr);
  brits_main<<<B_ / BM, 256, 0, stream>>>(values, masks, deltas, bgamma, breg, bih, bhh,
                                          den, wsWc, wsWg, wsWr, out, out + 1);
}
